// autoencoder_rnn3_15848429322984
// MI455X (gfx1250) — matrix operand order repaired and checked by reading, not compiled, not run on hardware
//
#include <hip/hip_runtime.h>
#include <hip/hip_bf16.h>
#include <math.h>

// ---------------------------------------------------------------------------
// Problem constants (from reference)
// ---------------------------------------------------------------------------
#define S_LEN 32
#define BATCH 8192
#define EDIM  64
#define HDIM  256     // encoder hidden == decoder hidden (H == H2)
#define G4H   1024    // 4*H
#define CDIM  128
#define LDIM  128
#define MDIM  128

typedef __attribute__((ext_vector_type(16))) __bf16 v16bf;
typedef __attribute__((ext_vector_type(8)))  float  v8f;
typedef unsigned short u16;
typedef unsigned int   u32;

// ---------------------------------------------------------------------------
// Helpers
// ---------------------------------------------------------------------------
__device__ __forceinline__ u16 f2bf(float f) {
    u32 u = __float_as_uint(f);
    u32 r = u + 0x7FFFu + ((u >> 16) & 1u);   // round-to-nearest-even
    return (u16)(r >> 16);
}

__device__ __forceinline__ float sigf(float x) {
    return 1.0f / (1.0f + __expf(-x));
}

// A-fragment (16x32, bf16): lane holds row m = lane%16.
// elements 0..7  = A[m, k0 + (lane/16)*8 + 0..7]
// elements 8..15 = A[m, k0 + 16 + (lane/16)*8 + 0..7]
__device__ __forceinline__ v16bf load_fragA(const u16* rowp, int k0, int kh) {
    union { uint4 u[2]; v16bf v; } t;
    const u16* p = rowp + k0 + kh * 8;
    t.u[0] = *(const uint4*)(p);
    t.u[1] = *(const uint4*)(p + 16);
    return t.v;
}

// B-fragment (32x16, bf16) built from W (N x K, row-major): lane holds col
// n = lane%16 -> W row n. elements 0..7 = W[n, k0 + (lane/16)*8 + 0..7],
// elements 8..15 = W[n, k0 + 16 + (lane/16)*8 + 0..7].
__device__ __forceinline__ v16bf load_fragB(const u16* rowp, int k0, int kh) {
    union { uint4 u[2]; v16bf v; } t;
    const u16* p = rowp + k0 + kh * 8;
    t.u[0] = *(const uint4*)(p);
    t.u[1] = *(const uint4*)(p + 16);
    return t.v;
}

// ---------------------------------------------------------------------------
// Weight / embedding prep kernels
// ---------------------------------------------------------------------------
__global__ void cvt_bf16_kernel(const float* __restrict__ src,
                                u16* __restrict__ dst, int n) {
    int i = blockIdx.x * blockDim.x + threadIdx.x;
    if (i < n) dst[i] = f2bf(src[i]);
}

__global__ void embed_kernel(const int* __restrict__ x,
                             const float* __restrict__ emb_kmers,
                             float* __restrict__ out_emb,
                             float* __restrict__ out_embr,
                             u16* __restrict__ emb_bf) {
    int i = blockIdx.x * blockDim.x + threadIdx.x;
    if (i >= S_LEN * BATCH * EDIM) return;
    int e  = i % EDIM;
    int sb = i / EDIM;
    int b  = sb % BATCH;
    int s  = sb / BATCH;
    int idx = x[s * BATCH + b];
    float v = emb_kmers[idx * EDIM + e];
    out_emb[i] = v;
    out_embr[((size_t)(S_LEN - 1 - s) * BATCH + b) * EDIM + e] = v;
    emb_bf[i] = f2bf(v);
}

// ---------------------------------------------------------------------------
// Fused LSTM step: gates = A1*W1^T + A2*W2^T + b1 + b2 ; apply cell update.
// One wave computes a 16(batch) x 16(hidden) tile and ALL FOUR gate tiles.
// ---------------------------------------------------------------------------
struct DirArgs {
    const u16*   a1;      // [BATCH x k1] bf16 (input)
    const u16*   w1;      // [1024 x k1]  bf16
    int          k1;      // 64 (encoder) or 256 (decoder)
    const u16*   a2;      // [BATCH x 256] bf16 (hidden state)
    const u16*   w2;      // [1024 x 256] bf16
    const float* b1;      // bih (1024)
    const float* b2;      // bhh (1024)
    float*       c;       // [BATCH x 256] fp32, in/out
    float*       h_out;   // [BATCH x 256] fp32 (ws buffer or d_out seq slot)
    u16*         hbf_out; // [BATCH x 256] bf16 (next-step state)
};

__global__ __launch_bounds__(256) void lstm_step_kernel(DirArgs fa, DirArgs ra) {
    DirArgs a = (blockIdx.z == 0) ? fa : ra;

    const int wave  = threadIdx.x >> 5;
    const int lane  = threadIdx.x & 31;
    const int gw    = blockIdx.x * 8 + wave;   // 8192 waves
    const int btile = gw >> 4;                 // 0..511
    const int nh    = gw & 15;                 // hidden tile 0..15
    const int col   = lane & 15;
    const int kh    = lane >> 4;
    const int hid   = nh * 16 + col;           // hidden unit 0..255

    v8f zero = {0.f, 0.f, 0.f, 0.f, 0.f, 0.f, 0.f, 0.f};
    v8f acc[4] = {zero, zero, zero, zero};     // i, f, g, o

    // Input projection: A1 (K = k1) x W1
    {
        const u16* a1row = a.a1 + (size_t)(btile * 16 + col) * a.k1;
        for (int k0 = 0; k0 < a.k1; k0 += 32) {
            v16bf af = load_fragA(a1row, k0, kh);
#pragma unroll
            for (int g = 0; g < 4; ++g) {
                const u16* wrow = a.w1 + (size_t)(g * HDIM + hid) * (size_t)a.k1;
                v16bf bf = load_fragB(wrow, k0, kh);
                acc[g] = __builtin_amdgcn_wmma_f32_16x16x32_bf16(
                    false, af, false, bf, (short)0, acc[g], false, false);
            }
        }
    }
    // Hidden projection: A2 (K = 256) x W2
    {
        const u16* a2row = a.a2 + (size_t)(btile * 16 + col) * HDIM;
        for (int k0 = 0; k0 < HDIM; k0 += 32) {
            v16bf af = load_fragA(a2row, k0, kh);
#pragma unroll
            for (int g = 0; g < 4; ++g) {
                const u16* wrow = a.w2 + (size_t)(g * HDIM + hid) * (size_t)HDIM;
                v16bf bf = load_fragB(wrow, k0, kh);
                acc[g] = __builtin_amdgcn_wmma_f32_16x16x32_bf16(
                    false, af, false, bf, (short)0, acc[g], false, false);
            }
        }
    }

    // LSTM cell epilogue (per-lane bias is uniform across the 8 acc elems)
    const float bI = a.b1[          hid] + a.b2[          hid];
    const float bF = a.b1[HDIM    + hid] + a.b2[HDIM    + hid];
    const float bG = a.b1[2*HDIM  + hid] + a.b2[2*HDIM  + hid];
    const float bO = a.b1[3*HDIM  + hid] + a.b2[3*HDIM  + hid];

#pragma unroll
    for (int v = 0; v < 8; ++v) {
        int m = btile * 16 + v + 8 * kh;       // batch row (C/D layout)
        size_t off = (size_t)m * HDIM + hid;
        float ig = sigf(acc[0][v] + bI);
        float fg = sigf(acc[1][v] + bF);
        float gg = tanhf(acc[2][v] + bG);
        float og = sigf(acc[3][v] + bO);
        float cn = fg * a.c[off] + ig * gg;
        float hn = og * tanhf(cn);
        a.c[off]       = cn;
        a.h_out[off]   = hn;
        a.hbf_out[off] = f2bf(hn);
    }
}

// ---------------------------------------------------------------------------
// Middle stages (small GEMMs, plain VALU)
// ---------------------------------------------------------------------------
__global__ void mid1_kernel(const float* __restrict__ hF, const float* __restrict__ hR,
                            const float* __restrict__ Wc,  const float* __restrict__ bc,
                            const float* __restrict__ Wlf, const float* __restrict__ blf,
                            const float* __restrict__ Wlr, const float* __restrict__ blr,
                            float* __restrict__ common,
                            float* __restrict__ out_lf, float* __restrict__ out_lr,
                            float* __restrict__ cF, float* __restrict__ cR,
                            u16* __restrict__ hbfF0, u16* __restrict__ hbfR0) {
    int i = blockIdx.x * blockDim.x + threadIdx.x;
    if (i >= BATCH * CDIM) return;
    int n = i % CDIM, b = i / CDIM;
    const float* hf = hF + (size_t)b * HDIM;
    const float* hr = hR + (size_t)b * HDIM;
    const float* wc  = Wc  + (size_t)n * (2 * HDIM);
    const float* wlf = Wlf + (size_t)n * HDIM;
    const float* wlr = Wlr + (size_t)n * HDIM;
    float ac = bc[n], alf = blf[n], alr = blr[n];
    for (int k = 0; k < HDIM; ++k) {
        float hfk = hf[k], hrk = hr[k];
        ac  += hfk * wc[k] + hrk * wc[HDIM + k];
        alf += hfk * wlf[k];
        alr += hrk * wlr[k];
    }
    common[i] = tanhf(ac);
    out_lf[i] = alf;
    out_lr[i] = alr;
    size_t off = (size_t)b * HDIM + n;   // h0 = [l, mix]; low half = l
    cF[off] = alf;  hbfF0[off] = f2bf(alf);
    cR[off] = alr;  hbfR0[off] = f2bf(alr);
}

__global__ void mid2_kernel(const int* __restrict__ Vg, const int* __restrict__ Jg,
                            const int* __restrict__ len_p,
                            const float* __restrict__ emb_Vg, const float* __restrict__ emb_Jg,
                            const float* __restrict__ common,
                            const float* __restrict__ Wmix, const float* __restrict__ bmix,
                            float* __restrict__ cF, float* __restrict__ cR,
                            u16* __restrict__ hbfF0, u16* __restrict__ hbfR0) {
    int i = blockIdx.x * blockDim.x + threadIdx.x;
    if (i >= BATCH * MDIM) return;
    int m = i % MDIM, b = i / MDIM;
    const float* wm = Wmix + (size_t)m * 133;
    int vg = Vg[b], jg = Jg[b];
    float acc = bmix[m];
    acc += emb_Vg[vg * 2 + 0] * wm[0] + emb_Vg[vg * 2 + 1] * wm[1];
    const float* cm = common + (size_t)b * CDIM;
    for (int j = 0; j < CDIM; ++j) acc += cm[j] * wm[2 + j];
    acc += emb_Jg[jg * 2 + 0] * wm[130] + emb_Jg[jg * 2 + 1] * wm[131];
    acc += (float)(*len_p) * wm[132];
    size_t off = (size_t)b * HDIM + CDIM + m;  // high half of h0 = mix (shared)
    cF[off] = acc;  cR[off] = acc;
    u16 bv = f2bf(acc);
    hbfF0[off] = bv;  hbfR0[off] = bv;
}

// start letters -> decoder first input (bf16) + seq endpoint writes
__global__ void mid3_kernel(const float* __restrict__ emb_letters,
                            u16* __restrict__ p0F, u16* __restrict__ p0R,
                            float* __restrict__ seqF, float* __restrict__ seqR) {
    int i = blockIdx.x * blockDim.x + threadIdx.x;
    if (i >= BATCH * HDIM) return;
    int n = i % HDIM;
    float sx = emb_letters[20 * HDIM + n];   // X_IDX
    float sb = emb_letters[21 * HDIM + n];   // B_IDX
    p0F[i] = f2bf(sx);
    p0R[i] = f2bf(sb);
    size_t last = (size_t)S_LEN * BATCH * HDIM + i;   // t = 32 (reversed start)
    seqF[last] = sx;  seqF[i] = sb;                    // t = 0  (reversed end)
    seqR[last] = sb;  seqR[i] = sx;
}

// ---------------------------------------------------------------------------
// Host-side orchestration
// ---------------------------------------------------------------------------
extern "C" void kernel_launch(void* const* d_in, const int* in_sizes, int n_in,
                              void* d_out, int out_size, void* d_ws, size_t ws_size,
                              hipStream_t stream) {
    (void)in_sizes; (void)n_in; (void)out_size; (void)ws_size;

    const int*   x        = (const int*)  d_in[0];
    const int*   Vg       = (const int*)  d_in[1];
    const int*   Jg       = (const int*)  d_in[2];
    const int*   len_p    = (const int*)  d_in[3];
    const float* emb_Vg   = (const float*)d_in[5];
    const float* emb_Jg   = (const float*)d_in[6];
    const float* emb_kmers= (const float*)d_in[7];
    const float* emb_let  = (const float*)d_in[8];
    const float* Wih_enc  = (const float*)d_in[9];
    const float* Whh_enc  = (const float*)d_in[10];
    const float* bih_enc  = (const float*)d_in[11];
    const float* bhh_enc  = (const float*)d_in[12];
    const float* Wc       = (const float*)d_in[13];
    const float* bc       = (const float*)d_in[14];
    const float* Wlf      = (const float*)d_in[15];
    const float* blf      = (const float*)d_in[16];
    const float* Wlr      = (const float*)d_in[17];
    const float* blr      = (const float*)d_in[18];
    const float* Wmix     = (const float*)d_in[19];
    const float* bmix     = (const float*)d_in[20];
    const float* Wih_f    = (const float*)d_in[21];
    const float* Whh_f    = (const float*)d_in[22];
    const float* bih_f    = (const float*)d_in[23];
    const float* bhh_f    = (const float*)d_in[24];
    const float* Wih_r    = (const float*)d_in[25];
    const float* Whh_r    = (const float*)d_in[26];
    const float* bih_r    = (const float*)d_in[27];
    const float* bhh_r    = (const float*)d_in[28];

    // ---- output layout (floats) -------------------------------------------
    float* out     = (float*)d_out;
    float* out_lf  = out;
    float* out_lr  = out + (size_t)BATCH * LDIM;
    float* seqF    = out + (size_t)2 * BATCH * LDIM;
    float* seqR    = seqF + (size_t)(S_LEN + 1) * BATCH * HDIM;
    float* out_emb = seqR + (size_t)(S_LEN + 1) * BATCH * HDIM;
    float* out_embr= out_emb + (size_t)S_LEN * BATCH * EDIM;

    // ---- workspace layout -------------------------------------------------
    char* ws = (char*)d_ws;
    size_t off = 0;
    auto alloc = [&](size_t bytes) -> void* {
        void* p = ws + off;
        off = (off + bytes + 255) & ~(size_t)255;
        return p;
    };
    u16*   emb_bf = (u16*)  alloc(sizeof(u16) * (size_t)S_LEN * BATCH * EDIM);
    u16*   wihE   = (u16*)  alloc(sizeof(u16) * G4H * EDIM);
    u16*   whhE   = (u16*)  alloc(sizeof(u16) * G4H * HDIM);
    u16*   wihFd  = (u16*)  alloc(sizeof(u16) * G4H * HDIM);
    u16*   whhFd  = (u16*)  alloc(sizeof(u16) * G4H * HDIM);
    u16*   wihRd  = (u16*)  alloc(sizeof(u16) * G4H * HDIM);
    u16*   whhRd  = (u16*)  alloc(sizeof(u16) * G4H * HDIM);
    float* hF     = (float*)alloc(sizeof(float) * (size_t)BATCH * HDIM);
    float* hR     = (float*)alloc(sizeof(float) * (size_t)BATCH * HDIM);
    float* cF     = (float*)alloc(sizeof(float) * (size_t)BATCH * HDIM);
    float* cR     = (float*)alloc(sizeof(float) * (size_t)BATCH * HDIM);
    u16*   hbfF[2], *hbfR[2];
    hbfF[0] = (u16*)alloc(sizeof(u16) * (size_t)BATCH * HDIM);
    hbfF[1] = (u16*)alloc(sizeof(u16) * (size_t)BATCH * HDIM);
    hbfR[0] = (u16*)alloc(sizeof(u16) * (size_t)BATCH * HDIM);
    hbfR[1] = (u16*)alloc(sizeof(u16) * (size_t)BATCH * HDIM);
    u16*   p0F    = (u16*)  alloc(sizeof(u16) * (size_t)BATCH * HDIM);
    u16*   p0R    = (u16*)  alloc(sizeof(u16) * (size_t)BATCH * HDIM);
    float* common = (float*)alloc(sizeof(float) * (size_t)BATCH * CDIM);

    // ---- init states ------------------------------------------------------
    hipMemsetAsync(cF, 0, sizeof(float) * (size_t)BATCH * HDIM, stream);
    hipMemsetAsync(cR, 0, sizeof(float) * (size_t)BATCH * HDIM, stream);
    hipMemsetAsync(hbfF[0], 0, sizeof(u16) * (size_t)BATCH * HDIM, stream);
    hipMemsetAsync(hbfR[0], 0, sizeof(u16) * (size_t)BATCH * HDIM, stream);

    // ---- weight conversion ------------------------------------------------
    auto cvt = [&](const float* s, u16* d, int n) {
        cvt_bf16_kernel<<<(n + 255) / 256, 256, 0, stream>>>(s, d, n);
    };
    cvt(Wih_enc, wihE, G4H * EDIM);
    cvt(Whh_enc, whhE, G4H * HDIM);
    cvt(Wih_f,   wihFd, G4H * HDIM);
    cvt(Whh_f,   whhFd, G4H * HDIM);
    cvt(Wih_r,   wihRd, G4H * HDIM);
    cvt(Whh_r,   whhRd, G4H * HDIM);

    // ---- embedding gather (also emits emb / emb_r outputs) ----------------
    {
        int n = S_LEN * BATCH * EDIM;
        embed_kernel<<<(n + 255) / 256, 256, 0, stream>>>(
            x, emb_kmers, out_emb, out_embr, emb_bf);
    }

    const dim3 lstm_grid(1024, 1, 2);   // 8192 waves per direction

    // ---- encoder: 32 steps, fwd + rev fused via grid.z --------------------
    for (int s = 0; s < S_LEN; ++s) {
        DirArgs fa = { emb_bf + (size_t)s * BATCH * EDIM,             wihE, EDIM,
                       hbfF[s & 1], whhE, bih_enc, bhh_enc,
                       cF, hF, hbfF[(s + 1) & 1] };
        DirArgs ra = { emb_bf + (size_t)(S_LEN - 1 - s) * BATCH * EDIM, wihE, EDIM,
                       hbfR[s & 1], whhE, bih_enc, bhh_enc,
                       cR, hR, hbfR[(s + 1) & 1] };
        lstm_step_kernel<<<lstm_grid, 256, 0, stream>>>(fa, ra);
    }

    // ---- middle stage -----------------------------------------------------
    {
        int n1 = BATCH * CDIM;
        mid1_kernel<<<(n1 + 255) / 256, 256, 0, stream>>>(
            hF, hR, Wc, bc, Wlf, blf, Wlr, blr,
            common, out_lf, out_lr, cF, cR, hbfF[0], hbfR[0]);
        mid2_kernel<<<(n1 + 255) / 256, 256, 0, stream>>>(
            Vg, Jg, len_p, emb_Vg, emb_Jg, common, Wmix, bmix,
            cF, cR, hbfF[0], hbfR[0]);
        int n3 = BATCH * HDIM;
        mid3_kernel<<<(n3 + 255) / 256, 256, 0, stream>>>(
            emb_let, p0F, p0R, seqF, seqR);
    }

    // ---- decoder: 31 steps; outs[j] lands at reversed index 31 - j --------
    for (int j = 0; j < S_LEN - 1; ++j) {
        const u16* pF = (j == 0) ? p0F : hbfF[j & 1];
        const u16* pR = (j == 0) ? p0R : hbfR[j & 1];
        float* hofF = seqF + (size_t)(S_LEN - 1 - j) * BATCH * HDIM;
        float* hofR = seqR + (size_t)(S_LEN - 1 - j) * BATCH * HDIM;
        DirArgs fa = { pF, wihFd, HDIM, hbfF[j & 1], whhFd, bih_f, bhh_f,
                       cF, hofF, hbfF[(j + 1) & 1] };
        DirArgs ra = { pR, wihRd, HDIM, hbfR[j & 1], whhRd, bih_r, bhh_r,
                       cR, hofR, hbfR[(j + 1) & 1] };
        lstm_step_kernel<<<lstm_grid, 256, 0, stream>>>(fa, ra);
    }
}
